// k1_GNN_sub_multi_h_sep_7842610283385
// MI455X (gfx1250) — matrix operand order repaired and checked by reading, not compiled, not run on hardware
//
#include <hip/hip_runtime.h>
#include <hip/hip_bf16.h>
#include <math.h>

// ---------------------------------------------------------------------------
// Problem constants (match reference)
// ---------------------------------------------------------------------------
static constexpr int N_ = 20000;   // nodes
static constexpr int E_ = 60000;   // edges (divisible by 16)
static constexpr int S_ = 3000;    // subgraphs
static constexpr int G_ = 128;     // graphs
static constexpr int HID = 128;    // edge-MLP hidden
static constexpr int XC = 24;      // raw node feature cols
static constexpr int FCF = 72;     // 64 + 8 pooled features

typedef __attribute__((ext_vector_type(16))) __bf16 v16bf;
typedef __attribute__((ext_vector_type(8)))  __bf16 v8bf;
typedef __attribute__((ext_vector_type(8)))  float  v8f;

__device__ __forceinline__ unsigned short f2bf(float f) {
  unsigned int u = __float_as_uint(f);
  u += 0x7FFFu + ((u >> 16) & 1u);          // round-to-nearest-even
  return (unsigned short)(u >> 16);
}
__device__ __forceinline__ float eluf(float v) {
  return v > 0.f ? v : (expf(v) - 1.f);
}

// ---------------------------------------------------------------------------
// 1) Edge MLP: H[e][k] = relu(ea[e] @ w1 + b1)  -> bf16, row-major E x 128
// ---------------------------------------------------------------------------
__global__ void edge_h_kernel(const float* __restrict__ ea,
                              const float* __restrict__ w1,
                              const float* __restrict__ b1,
                              unsigned short* __restrict__ Hbf) {
  int idx = blockIdx.x * blockDim.x + threadIdx.x;
  if (idx >= E_ * HID) return;
  int e = idx >> 7, k = idx & 127;
  float v = b1[k];
#pragma unroll
  for (int d = 0; d < 5; ++d) v = fmaf(ea[e * 5 + d], w1[d * HID + k], v);
  v = v > 0.f ? v : 0.f;
  Hbf[idx] = f2bf(v);
}

// ---------------------------------------------------------------------------
// 2) Pack w2 (128 x P fp32, P = mi*mo) into wave32 WMMA B-fragment layout:
//    Bp[((kk*(P/16) + ob)*32 + lane)*16 + j]  (bf16)
//    lane<16 holds K = kk*32 + j, lane>=16 holds K = kk*32 + 16 + j,
//    column = ob*16 + (lane&15). One 32B contiguous load per lane per WMMA.
// ---------------------------------------------------------------------------
__global__ void pack_b_kernel(const float* __restrict__ w2,
                              unsigned short* __restrict__ Bp, int P) {
  int idx = blockIdx.x * blockDim.x + threadIdx.x;
  int total = 4 * (P / 16) * 32;
  if (idx >= total) return;
  int lane  = idx & 31;
  int blk   = idx >> 5;                 // kk*(P/16) + ob
  int ob    = blk % (P / 16);
  int kk    = blk / (P / 16);
  int col   = ob * 16 + (lane & 15);
  int khalf = lane >> 4;
  unsigned short* out = Bp + (size_t)idx * 16;
#pragma unroll
  for (int j = 0; j < 16; ++j) {
    int k = kk * 32 + khalf * 16 + j;
    out[j] = f2bf(w2[(size_t)k * P + col]);
  }
}

// ---------------------------------------------------------------------------
// 3) Fused  W = H @ w2 + b2  ->  msg = einsum('ei,eio->eo')  -> atomic scatter
//    64 threads (2 waves) per 16-edge tile. bf16 WMMA, fp32 accumulate.
// ---------------------------------------------------------------------------
template <int MI, int MO>
__launch_bounds__(64)
__global__ void conv_msg_wmma(const unsigned short* __restrict__ Hbf,
                              const unsigned short* __restrict__ Bp,
                              const float* __restrict__ b2,
                              const float* __restrict__ xin, int ldx,
                              const int* __restrict__ srcIdx,
                              const int* __restrict__ dstIdx,
                              float* __restrict__ agg) {
  constexpr int P   = MI * MO;
  constexpr int NOB = MO / 16;
  __shared__ __attribute__((aligned(16))) unsigned short sH[16 * HID];
  __shared__ float sX[16][MI];
  __shared__ int   sDst[16];

  const int tid = threadIdx.x;
  const int e0  = blockIdx.x * 16;

  // stage H tile (16 x 128 bf16 = 4KB) via b128 loads
  {
    const uint4* g = (const uint4*)(Hbf + (size_t)e0 * HID);
    uint4* s = (uint4*)sH;
#pragma unroll
    for (int j = 0; j < 4; ++j) s[tid + j * 64] = g[tid + j * 64];
  }
  if (tid < 16) sDst[tid] = dstIdx[e0 + tid];
  for (int t = tid; t < 16 * MI; t += 64) {
    int r = t / MI, i = t % MI;
    sX[r][i] = xin[(size_t)srcIdx[e0 + r] * ldx + i];
  }
  __syncthreads();

  const int wave = tid >> 5;
  const int lane = tid & 31;
  const int l15  = lane & 15;
  const int hi   = lane >> 4;

  for (int ob = wave; ob < NOB; ob += 2) {
    v8f msgacc;
#pragma unroll
    for (int r = 0; r < 8; ++r) msgacc[r] = 0.f;

    for (int i = 0; i < MI; ++i) {
      const int cblk = i * MO + ob * 16;     // column block base in P space
      const float b2v = b2[cblk + l15];
      v8f c;
#pragma unroll
      for (int r = 0; r < 8; ++r) c[r] = b2v;

      const unsigned short* bp = Bp + (size_t)((cblk >> 4) * 32 + lane) * 16;
      if (i + 1 < MI)  // warm L0 for the next i's B-fragments
        __builtin_prefetch(Bp + (size_t)(((cblk + MO) >> 4) * 32 + lane) * 16, 0, 1);

#pragma unroll
      for (int kk = 0; kk < 4; ++kk) {
        union { v16bf v; v8bf h[2]; } a;
        a.h[0] = *(const v8bf*)&sH[l15 * HID + kk * 32 + hi * 16];
        a.h[1] = *(const v8bf*)&sH[l15 * HID + kk * 32 + hi * 16 + 8];
        const v16bf bv = *(const v16bf*)(bp + (size_t)kk * (P / 16) * 32 * 16);
        c = __builtin_amdgcn_wmma_f32_16x16x32_bf16(
                false, a.v, false, bv, (short)0, c, false, false);
      }
      // msg[e, o] += x[src[e], i] * W[e, i*MO + o]; C row M = r + 8*hi
#pragma unroll
      for (int r = 0; r < 8; ++r)
        msgacc[r] = fmaf(sX[r + 8 * hi][i], c[r], msgacc[r]);
    }
    // scatter-add into destination nodes
#pragma unroll
    for (int r = 0; r < 8; ++r)
      atomicAdd(&agg[(size_t)sDst[r + 8 * hi] * MO + ob * 16 + l15], msgacc[r]);
  }
}

// ---------------------------------------------------------------------------
// 4) out = elu(agg + x @ root + bias)
// ---------------------------------------------------------------------------
__global__ void conv_root_kernel(const float* __restrict__ agg,
                                 const float* __restrict__ xin, int ldx,
                                 const float* __restrict__ root,
                                 const float* __restrict__ bias,
                                 float* __restrict__ xout, int MI, int MO) {
  int idx = blockIdx.x * blockDim.x + threadIdx.x;
  if (idx >= N_ * MO) return;
  int n = idx / MO, o = idx % MO;
  float v = agg[idx] + bias[o];
  for (int i = 0; i < MI; ++i)
    v = fmaf(xin[(size_t)n * ldx + i], root[i * MO + o], v);
  xout[idx] = eluf(v);
}

// ---------------------------------------------------------------------------
// 5) Pooling chain
// ---------------------------------------------------------------------------
__global__ void concat_pool1_kernel(const float* __restrict__ xi,
                                    const float* __restrict__ x,
                                    const int* __restrict__ n2s,
                                    float* __restrict__ ssum,
                                    float* __restrict__ scnt) {
  int idx = blockIdx.x * blockDim.x + threadIdx.x;
  if (idx >= N_ * FCF) return;
  int n = idx / FCF, f = idx % FCF;
  float v = (f < 64) ? xi[(size_t)n * 64 + f]
                     : x[(size_t)n * XC + 16 + (f - 64)];
  int s = n2s[n];
  atomicAdd(&ssum[(size_t)s * FCF + f], v);
  if (f == 0) atomicAdd(&scnt[s], 1.0f);
}

__global__ void mean_div_kernel(float* __restrict__ out,
                                const float* __restrict__ sum,
                                const float* __restrict__ cnt, int total) {
  int idx = blockIdx.x * blockDim.x + threadIdx.x;
  if (idx >= total) return;
  out[idx] = sum[idx] / fmaxf(cnt[idx / FCF], 1.0f);
}

__global__ void pool2_kernel(const float* __restrict__ xsub,
                             const int* __restrict__ s2g,
                             float* __restrict__ gsum,
                             float* __restrict__ gcnt) {
  int idx = blockIdx.x * blockDim.x + threadIdx.x;
  if (idx >= S_ * FCF) return;
  int s = idx / FCF, f = idx % FCF;
  int g = s2g[s];
  atomicAdd(&gsum[(size_t)g * FCF + f], xsub[idx]);
  if (f == 0) atomicAdd(&gcnt[g], 1.0f);
}

// ---------------------------------------------------------------------------
// 6) FC head: 72 -> 36 -> 18 -> 1, one thread per graph
// ---------------------------------------------------------------------------
__global__ void fc_head_kernel(const float* __restrict__ xg,
                               const float* __restrict__ w1, const float* __restrict__ b1,
                               const float* __restrict__ w2, const float* __restrict__ b2,
                               const float* __restrict__ w3, const float* __restrict__ b3,
                               float* __restrict__ out) {
  int g = blockIdx.x * blockDim.x + threadIdx.x;
  if (g >= G_) return;
  const float* in = xg + (size_t)g * FCF;
  float h1[36];
  for (int j = 0; j < 36; ++j) {
    float v = b1[j];
    for (int i = 0; i < FCF; ++i) v = fmaf(in[i], w1[i * 36 + j], v);
    h1[j] = eluf(v);
  }
  float h2[18];
  for (int j = 0; j < 18; ++j) {
    float v = b2[j];
    for (int i = 0; i < 36; ++i) v = fmaf(h1[i], w2[i * 18 + j], v);
    h2[j] = eluf(v);
  }
  float v = b3[0];
  for (int i = 0; i < 18; ++i) v = fmaf(h2[i], w3[i], v);
  out[g] = v;
}

// ---------------------------------------------------------------------------
// Host launcher
// ---------------------------------------------------------------------------
static inline size_t align256(size_t x) { return (x + 255) & ~(size_t)255; }

extern "C" void kernel_launch(void* const* d_in, const int* in_sizes, int n_in,
                              void* d_out, int out_size, void* d_ws, size_t ws_size,
                              hipStream_t stream) {
  (void)in_sizes; (void)n_in; (void)out_size; (void)ws_size;

  const float* x   = (const float*)d_in[0];
  const int*   ei  = (const int*)d_in[1];
  const float* ea  = (const float*)d_in[2];
  const int*   n2s = (const int*)d_in[3];
  const int*   s2g = (const int*)d_in[4];
  const int* src = ei;
  const int* dst = ei + E_;

  const float* cw1[3], * cb1[3], * cw2[3], * cb2[3], * crt[3], * cbs[3];
  for (int L = 0; L < 3; ++L) {
    int b = 5 + 6 * L;
    cw1[L] = (const float*)d_in[b + 0];
    cb1[L] = (const float*)d_in[b + 1];
    cw2[L] = (const float*)d_in[b + 2];
    cb2[L] = (const float*)d_in[b + 3];
    crt[L] = (const float*)d_in[b + 4];
    cbs[L] = (const float*)d_in[b + 5];
  }
  const float* fc1w = (const float*)d_in[23];
  const float* fc1b = (const float*)d_in[24];
  const float* fc2w = (const float*)d_in[25];
  const float* fc2b = (const float*)d_in[26];
  const float* fc3w = (const float*)d_in[27];
  const float* fc3b = (const float*)d_in[28];

  // workspace carve-up
  char* base = (char*)d_ws;
  size_t off = 0;
  auto carve = [&](size_t bytes) { void* p = base + off; off += align256(bytes); return p; };
  unsigned short* Hbf = (unsigned short*)carve((size_t)E_ * HID * 2);
  unsigned short* Bp  = (unsigned short*)carve((size_t)HID * 4096 * 2);
  float* agg  = (float*)carve((size_t)N_ * 64 * 4);
  float* xiA  = (float*)carve((size_t)N_ * 64 * 4);
  float* xiB  = (float*)carve((size_t)N_ * 64 * 4);
  float* ssum = (float*)carve((size_t)S_ * FCF * 4);
  float* scnt = (float*)carve((size_t)S_ * 4);
  float* xsub = (float*)carve((size_t)S_ * FCF * 4);
  float* gsum = (float*)carve((size_t)G_ * FCF * 4);
  float* gcnt = (float*)carve((size_t)G_ * 4);
  float* xg   = (float*)carve((size_t)G_ * FCF * 4);

  const int TB = 256;
  const int ehBlocks = (E_ * HID + TB - 1) / TB;

  // ---- Layer 0: MI=16 (x[:, :16], ld 24) -> xiA (MO=32) ----
  {
    constexpr int MI = 16, MO = 32, P = MI * MO;
    edge_h_kernel<<<ehBlocks, TB, 0, stream>>>(ea, cw1[0], cb1[0], Hbf);
    int tot = 4 * (P / 16) * 32;
    pack_b_kernel<<<(tot + 127) / 128, 128, 0, stream>>>(cw2[0], Bp, P);
    hipMemsetAsync(agg, 0, (size_t)N_ * MO * 4, stream);
    conv_msg_wmma<MI, MO><<<E_ / 16, 64, 0, stream>>>(Hbf, Bp, cb2[0], x, XC, src, dst, agg);
    conv_root_kernel<<<(N_ * MO + TB - 1) / TB, TB, 0, stream>>>(agg, x, XC, crt[0], cbs[0], xiA, MI, MO);
  }
  // ---- Layer 1: xiA (ld 32) -> xiB (MO=64) ----
  {
    constexpr int MI = 32, MO = 64, P = MI * MO;
    edge_h_kernel<<<ehBlocks, TB, 0, stream>>>(ea, cw1[1], cb1[1], Hbf);
    int tot = 4 * (P / 16) * 32;
    pack_b_kernel<<<(tot + 127) / 128, 128, 0, stream>>>(cw2[1], Bp, P);
    hipMemsetAsync(agg, 0, (size_t)N_ * MO * 4, stream);
    conv_msg_wmma<MI, MO><<<E_ / 16, 64, 0, stream>>>(Hbf, Bp, cb2[1], xiA, MI, src, dst, agg);
    conv_root_kernel<<<(N_ * MO + TB - 1) / TB, TB, 0, stream>>>(agg, xiA, MI, crt[1], cbs[1], xiB, MI, MO);
  }
  // ---- Layer 2: xiB (ld 64) -> xiA (MO=64) ----
  {
    constexpr int MI = 64, MO = 64, P = MI * MO;
    edge_h_kernel<<<ehBlocks, TB, 0, stream>>>(ea, cw1[2], cb1[2], Hbf);
    int tot = 4 * (P / 16) * 32;
    pack_b_kernel<<<(tot + 127) / 128, 128, 0, stream>>>(cw2[2], Bp, P);
    hipMemsetAsync(agg, 0, (size_t)N_ * MO * 4, stream);
    conv_msg_wmma<MI, MO><<<E_ / 16, 64, 0, stream>>>(Hbf, Bp, cb2[2], xiB, MI, src, dst, agg);
    conv_root_kernel<<<(N_ * MO + TB - 1) / TB, TB, 0, stream>>>(agg, xiB, MI, crt[2], cbs[2], xiA, MI, MO);
  }

  // ---- concat + mean pools ----
  hipMemsetAsync(ssum, 0, (size_t)S_ * FCF * 4, stream);
  hipMemsetAsync(scnt, 0, (size_t)S_ * 4, stream);
  hipMemsetAsync(gsum, 0, (size_t)G_ * FCF * 4, stream);
  hipMemsetAsync(gcnt, 0, (size_t)G_ * 4, stream);
  concat_pool1_kernel<<<(N_ * FCF + TB - 1) / TB, TB, 0, stream>>>(xiA, x, n2s, ssum, scnt);
  mean_div_kernel<<<(S_ * FCF + TB - 1) / TB, TB, 0, stream>>>(xsub, ssum, scnt, S_ * FCF);
  pool2_kernel<<<(S_ * FCF + TB - 1) / TB, TB, 0, stream>>>(xsub, s2g, gsum, gcnt);
  mean_div_kernel<<<(G_ * FCF + TB - 1) / TB, TB, 0, stream>>>(xg, gsum, gcnt, G_ * FCF);

  // ---- FC head ----
  fc_head_kernel<<<1, G_, 0, stream>>>(xg, fc1w, fc1b, fc2w, fc2b, fc3w, fc3b, (float*)d_out);
}
